// SlotAttention_22557168238748
// MI455X (gfx1250) — matrix operand order repaired and checked by reading, not compiled, not run on hardware
//
#include <hip/hip_runtime.h>

// ---------------------------------------------------------------------------
// Slot-attention with MESH/Sinkhorn for MI455X (gfx1250, wave32, WMMA).
// Strategy:
//   * bf16 WMMA (v_wmma_f32_16x16x32_bf16) for all GEMMs: x@Wk, x@Wv,
//     the cdist gram k.q, and updates = T^T @ v.
//   * One workgroup (512 thr = 16 wave32) per batch keeps the whole
//     (1024 x 16) cost matrix + grad buffer + dual history in LDS
//     (~185 KB of the 320 KB/WGP) for the entire MESH pipeline.
//   * Hand-derived VJP through 5 Sinkhorn iterations (4 MESH steps),
//     warm-started duals, then final Sinkhorn, GRU + MLP fused.
// ---------------------------------------------------------------------------

typedef __bf16 bf16;
typedef __attribute__((ext_vector_type(16))) __bf16 v16bf;
typedef __attribute__((ext_vector_type(8)))  float  v8f;

#define B_    256
#define N_    1024
#define D_    64
#define S_    11
#define H_    128
#define ITERS_      3
#define MESH_ITERS_ 4

__device__ __forceinline__ v8f wmma_bf16(v16bf a, v16bf b, v8f c) {
  return __builtin_amdgcn_wmma_f32_16x16x32_bf16(false, a, false, b, (short)0, c,
                                                 false, false);
}

// A-operand fragment: 16(M) x 32(K), source row-major [m][k].
// ISA 7.12.2 16-bit A: lane(m=l&15,h=l>>4); vgpr j<4 -> K=8h+2j ; j>=4 -> 16+8h+2(j-4)
template <typename T>
__device__ __forceinline__ v16bf frag_a(const T* base, int stride, int koff) {
  const int lane = threadIdx.x & 31;
  const int m = lane & 15, h = lane >> 4;
  v16bf f;
#pragma unroll
  for (int j = 0; j < 8; ++j) {
    const int kb = koff + ((j < 4) ? (8 * h + 2 * j) : (16 + 8 * h + 2 * (j - 4)));
    f[2 * j]     = (__bf16)base[m * stride + kb];
    f[2 * j + 1] = (__bf16)base[m * stride + kb + 1];
  }
  return f;
}

// B-operand fragment: 32(K) x 16(N), source stored as B^T row-major [n][k].
// 16-bit B (matches A): lane(n=l&15,h=l>>4); vgpr j<4 -> K=8h+2j ; j>=4 -> 16+8h+2(j-4)
template <typename T>
__device__ __forceinline__ v16bf frag_b(const T* base, int stride, int koff) {
  const int lane = threadIdx.x & 31;
  const int n = lane & 15, h = lane >> 4;
  v16bf f;
#pragma unroll
  for (int j = 0; j < 8; ++j) {
    const int kb = koff + ((j < 4) ? (8 * h + 2 * j) : (16 + 8 * h + 2 * (j - 4)));
    f[2 * j]     = (__bf16)base[n * stride + kb];
    f[2 * j + 1] = (__bf16)base[n * stride + kb + 1];
  }
  return f;
}

__device__ __forceinline__ float wave_sum(float x) {
#pragma unroll
  for (int off = 16; off > 0; off >>= 1) x += __shfl_xor(x, off, 32);
  return x;
}
__device__ __forceinline__ float wave_max(float x) {
#pragma unroll
  for (int off = 16; off > 0; off >>= 1) x = fmaxf(x, __shfl_xor(x, off, 32));
  return x;
}

// ---------------- projection kernel: LN(x), k = x@Wk, v = x@Wv, wi logits ----
__global__ __launch_bounds__(256) void proj_kernel(
    const float* __restrict__ x, const float* __restrict__ Wk,
    const float* __restrict__ Wv, const float* __restrict__ lng,
    const float* __restrict__ lnb, const float* __restrict__ wi_w,
    const float* __restrict__ wi_b, bf16* __restrict__ k_bf,
    bf16* __restrict__ v_bf, float* __restrict__ logits) {
  extern __shared__ float shp[];
  float* xs = shp;                     // 128 x 66 f32 (LN'd x)
  bf16* wkt = (bf16*)(shp + 128 * 66); // 64 x 66 bf16 (Wk^T)
  bf16* wvt = wkt + 64 * 66;           // 64 x 66 bf16 (Wv^T)
  const int tid = threadIdx.x;
  const size_t r0 = (size_t)blockIdx.x * 128;

  for (int j = tid; j < 128 * 64; j += 256) {
    int r = j >> 6, c = j & 63;
    xs[r * 66 + c] = x[(r0 + r) * 64 + c];
  }
  for (int j = tid; j < 64 * 64; j += 256) {
    int din = j >> 6, dout = j & 63;
    wkt[dout * 66 + din] = (bf16)Wk[j];
    wvt[dout * 66 + din] = (bf16)Wv[j];
  }
  __syncthreads();
  if (tid < 128) {
    float* row = xs + tid * 66;
    float m = 0.f;
    for (int c = 0; c < 64; ++c) m += row[c];
    m *= (1.f / 64.f);
    float v = 0.f;
    for (int c = 0; c < 64; ++c) { float d = row[c] - m; v += d * d; }
    v *= (1.f / 64.f);
    float inv = rsqrtf(v + 1e-5f);
    float lg = wi_b[0];
    for (int c = 0; c < 64; ++c) {
      float xn = (row[c] - m) * inv * lng[c] + lnb[c];
      row[c] = xn;
      lg += xn * wi_w[c];
    }
    logits[r0 + tid] = lg;
  }
  __syncthreads();
  const int w = tid >> 5, lane = tid & 31;
  const int ncol = lane & 15, h = lane >> 4;
  v16bf a0 = frag_a(xs + w * 16 * 66, 66, 0);
  v16bf a1 = frag_a(xs + w * 16 * 66, 66, 32);
#pragma unroll
  for (int ctile = 0; ctile < 4; ++ctile) {
    v8f acck = {};
    acck = wmma_bf16(a0, frag_b(wkt + ctile * 16 * 66, 66, 0), acck);
    acck = wmma_bf16(a1, frag_b(wkt + ctile * 16 * 66, 66, 32), acck);
    v8f accv = {};
    accv = wmma_bf16(a0, frag_b(wvt + ctile * 16 * 66, 66, 0), accv);
    accv = wmma_bf16(a1, frag_b(wvt + ctile * 16 * 66, 66, 32), accv);
#pragma unroll
    for (int r = 0; r < 8; ++r) {
      size_t rowg = r0 + w * 16 + r + 8 * h;
      k_bf[rowg * 64 + ctile * 16 + ncol] = (bf16)acck[r];
      v_bf[rowg * 64 + ctile * 16 + ncol] = (bf16)accv[r];
    }
  }
}

// ---------------- k2 = sum(k*k, -1) ------------------------------------------
__global__ __launch_bounds__(256) void k2_kernel(const bf16* __restrict__ k_bf,
                                                 float* __restrict__ k2) {
  const size_t row = (size_t)blockIdx.x * 256 + threadIdx.x;  // < B*N
  const bf16* r = k_bf + row * 64;
  float s = 0.f;
  for (int d = 0; d < 64; ++d) { float f = (float)r[d]; s += f * f; }
  k2[row] = s;
}

// ---------------- log_a = log(softmax(logits)*S) -----------------------------
__global__ __launch_bounds__(256) void softmax_a_kernel(
    const float* __restrict__ logits, float* __restrict__ log_a) {
  __shared__ float red[256];
  const int b = blockIdx.x, tid = threadIdx.x;
  const float* lg = logits + (size_t)b * N_;
  float mx = -3.4e38f;
  for (int n = tid; n < N_; n += 256) mx = fmaxf(mx, lg[n]);
  red[tid] = mx;
  __syncthreads();
  for (int s = 128; s > 0; s >>= 1) {
    if (tid < s) red[tid] = fmaxf(red[tid], red[tid + s]);
    __syncthreads();
  }
  mx = red[0];
  __syncthreads();
  float sm = 0.f;
  for (int n = tid; n < N_; n += 256) sm += __expf(lg[n] - mx);
  red[tid] = sm;
  __syncthreads();
  for (int s = 128; s > 0; s >>= 1) {
    if (tid < s) red[tid] += red[tid + s];
    __syncthreads();
  }
  const float lse = mx + __logf(red[0]);
  const float lS = __logf((float)S_);
  for (int n = tid; n < N_; n += 256)
    log_a[(size_t)b * N_ + n] = lg[n] - lse + lS;
}

// ---------------- slots = mu + exp(log_sigma) * init -------------------------
__global__ void slots_init_kernel(const float* __restrict__ si,
                                  const float* __restrict__ mu,
                                  const float* __restrict__ lsig,
                                  float* __restrict__ slots) {
  int i = blockIdx.x * 256 + threadIdx.x;
  if (i < B_ * S_ * D_) {
    int d = i & 63;
    slots[i] = mu[d] + __expf(lsig[d]) * si[i];
  }
}

__global__ void copy_kernel(const float* __restrict__ src, float* __restrict__ dst) {
  int i = blockIdx.x * 256 + threadIdx.x;
  if (i < B_ * S_ * D_) dst[i] = src[i];
}

// ---------------- iteration kernel LDS layout (floats) -----------------------
constexpr int SH_P    = 0;                   // 1024x16  p = -C
constexpr int SH_GP   = SH_P + 16384;        // 1024x16  grad / T
constexpr int SH_UH   = SH_GP + 16384;       // 5x1024   u history (+GRU scratch)
constexpr int SH_GU   = SH_UH + 5120;        // 1024
constexpr int SH_LA   = SH_GU + 1024;        // 1024     log_a slice
constexpr int SH_SNB  = SH_LA + 1024;        // 16x66    LN(slots)
constexpr int SH_QB   = SH_SNB + 1056;       // 16x66    q (later ff-LN)
constexpr int SH_SL   = SH_QB + 1056;        // 16x64    slots local
constexpr int SH_UPD  = SH_SL + 1024;        // 16x64    updates
constexpr int SH_STGT = SH_UPD + 1024;       // 16x34    T^T stage
constexpr int SH_STGV = SH_STGT + 544;       // 64x34    v^T stage
constexpr int SH_VH   = SH_STGV + 2176;      // 6x16     v history
constexpr int SH_GV   = SH_VH + 96;          // 16
constexpr int SH_LOGB = SH_GV + 16;          // 16
constexpr int SH_Q2   = SH_LOGB + 16;        // 16
constexpr int SH_LS   = SH_Q2 + 16;          // 16
constexpr int SH_RED  = SH_LS + 16;          // 16x16 wave partials
constexpr int SH_WT   = SH_RED + 256;        // 16
constexpr int SH_TOT  = SH_WT + 16;          // 47248 floats = 188,992 B

// 5-iteration log-domain Sinkhorn forward; stores u_t (t=1..5) and v_t (t=1..5)
__device__ __forceinline__ void sk_forward(float* p, float* uh, float* vh,
                                           const float* la, const float* logb) {
  const int tid = threadIdx.x;
  const int w = tid >> 5, lane = tid & 31;
  for (int t = 1; t <= 5; ++t) {
    const float* vc = vh + (t - 1) * 16;
    float* uc = uh + (t - 1) * 1024;
    for (int n = tid; n < N_; n += 512) {
      const float* pr = p + n * 16;
      float M = -3.4e38f;
#pragma unroll
      for (int s = 0; s < 16; ++s) M = fmaxf(M, pr[s] + vc[s]);
      float sm = 0.f;
#pragma unroll
      for (int s = 0; s < 16; ++s) sm += __expf(pr[s] + vc[s] - M);
      uc[n] = la[n] - (M + __logf(sm));
    }
    __syncthreads();
    {  // one wave per slot column: LSE over n
      const int s = w;
      float M = -3.4e38f;
      for (int n = lane; n < N_; n += 32) M = fmaxf(M, p[n * 16 + s] + uc[n]);
      M = wave_max(M);
      float sm = 0.f;
      for (int n = lane; n < N_; n += 32) sm += __expf(p[n * 16 + s] + uc[n] - M);
      sm = wave_sum(sm);
      if (lane == 0) vh[t * 16 + s] = logb[s] - (M + __logf(sm));
    }
    __syncthreads();
  }
}

// VJP of H(T(sinkhorn)) w.r.t. p ; accumulates into gp (assigned at seed)
__device__ __forceinline__ void mesh_backward(float* p, float* gp, float* uh,
                                              float* vh, float* gu, float* gv,
                                              float* Ls, float* red) {
  const int tid = threadIdx.x;
  const int w = tid >> 5, lane = tid & 31;
  {  // seed: T, dH/dT, gp = dT*T ; gu = rowsum ; gv = colsum
    const float* u5 = uh + 4 * 1024;
    const float* v5 = vh + 5 * 16;
    float gvp[16];
#pragma unroll
    for (int s = 0; s < 16; ++s) gvp[s] = 0.f;
    for (int n = tid; n < N_; n += 512) {
      const float* pr = p + n * 16;
      float* gr = gp + n * 16;
      const float u = u5[n];
      float rowacc = 0.f;
#pragma unroll
      for (int s = 0; s < 16; ++s) {
        float T = __expf(pr[s] + u + v5[s]);
        float Te = T + 1e-8f;
        float dT = -(__logf(Te) + T / Te);
        float g = dT * T;
        gr[s] = g;
        rowacc += g;
        gvp[s] += g;
      }
      gu[n] = rowacc;
    }
#pragma unroll
    for (int s = 0; s < 16; ++s) {
      float xx = wave_sum(gvp[s]);
      if (lane == 0) red[w * 16 + s] = xx;
    }
    __syncthreads();
    if (tid < 16) {
      float xx = 0.f;
      for (int ww = 0; ww < 16; ++ww) xx += red[ww * 16 + tid];
      gv[tid] = xx;
    }
    __syncthreads();
  }
  for (int t = 5; t >= 1; --t) {
    const float* ut = uh + (t - 1) * 1024;
    const float* vtm1 = vh + (t - 1) * 16;
    {  // Ls[s] = LSE_n(p + u_t) (softmax denominator of the v-update)
      const int s = w;
      float M = -3.4e38f;
      for (int n = lane; n < N_; n += 32) M = fmaxf(M, p[n * 16 + s] + ut[n]);
      M = wave_max(M);
      float sm = 0.f;
      for (int n = lane; n < N_; n += 32) sm += __expf(p[n * 16 + s] + ut[n] - M);
      sm = wave_sum(sm);
      if (lane == 0) Ls[s] = M + __logf(sm);
    }
    __syncthreads();
    // v-update adjoint: gp += -gv*w ; gu += -sum_s gv*w
    for (int n = tid; n < N_; n += 512) {
      const float* pr = p + n * 16;
      float* gr = gp + n * 16;
      const float u = ut[n];
      float acc = 0.f;
#pragma unroll
      for (int s = 0; s < 16; ++s) {
        float wv = __expf(pr[s] + u - Ls[s]);
        float c = -gv[s] * wv;
        gr[s] += c;
        acc += c;
      }
      gu[n] += acc;
    }
    __syncthreads();
    // u-update adjoint: gp += -gu*m ; gv_{t-1} = -sum_n gu*m
    float gvp[16];
#pragma unroll
    for (int s = 0; s < 16; ++s) gvp[s] = 0.f;
    for (int n = tid; n < N_; n += 512) {
      const float* pr = p + n * 16;
      float* gr = gp + n * 16;
      float M = -3.4e38f;
#pragma unroll
      for (int s = 0; s < 16; ++s) M = fmaxf(M, pr[s] + vtm1[s]);
      float e[16];
      float sm = 0.f;
#pragma unroll
      for (int s = 0; s < 16; ++s) {
        e[s] = __expf(pr[s] + vtm1[s] - M);
        sm += e[s];
      }
      const float inv = 1.f / sm;
      const float g = gu[n];
#pragma unroll
      for (int s = 0; s < 16; ++s) {
        float c = -g * e[s] * inv;
        gr[s] += c;
        gvp[s] += c;
      }
      gu[n] = 0.f;
    }
#pragma unroll
    for (int s = 0; s < 16; ++s) {
      float xx = wave_sum(gvp[s]);
      if (lane == 0) red[w * 16 + s] = xx;
    }
    __syncthreads();
    if (tid < 16) {
      float xx = 0.f;
      for (int ww = 0; ww < 16; ++ww) xx += red[ww * 16 + tid];
      gv[tid] = xx;
    }
    __syncthreads();
  }
}

// ---------------- one slot-attention iteration, one workgroup per batch ------
__global__ __launch_bounds__(512) void iter_kernel(
    const bf16* __restrict__ k_bf, const bf16* __restrict__ v_bf,
    const float* __restrict__ k2g, const float* __restrict__ la_g,
    float* __restrict__ slots_g, float* __restrict__ attn_out,
    const float* __restrict__ Wq, const float* __restrict__ lnsg,
    const float* __restrict__ lnsb, const float* __restrict__ ws_w,
    const float* __restrict__ ws_b, const float* __restrict__ Wih,
    const float* __restrict__ Whh, const float* __restrict__ bih,
    const float* __restrict__ bhh, const float* __restrict__ fc1_w,
    const float* __restrict__ fc1_b, const float* __restrict__ fc2_w,
    const float* __restrict__ fc2_b, const float* __restrict__ lnfg,
    const float* __restrict__ lnfb) {
  extern __shared__ float sh[];
  float* p = sh + SH_P;
  float* gp = sh + SH_GP;
  float* uh = sh + SH_UH;
  float* gu = sh + SH_GU;
  float* la = sh + SH_LA;
  float* snb = sh + SH_SNB;
  float* qb = sh + SH_QB;
  float* sl = sh + SH_SL;
  float* upd = sh + SH_UPD;
  float* stgT = sh + SH_STGT;
  float* stgV = sh + SH_STGV;
  float* vh = sh + SH_VH;
  float* gv = sh + SH_GV;
  float* logb = sh + SH_LOGB;
  float* q2 = sh + SH_Q2;
  float* Ls = sh + SH_LS;
  float* red = sh + SH_RED;
  float* wt = sh + SH_WT;

  const int tid = threadIdx.x;
  const int w = tid >> 5, lane = tid & 31;
  const int b = blockIdx.x;
  const bf16* kbg = k_bf + (size_t)b * N_ * D_;
  const bf16* vbg = v_bf + (size_t)b * N_ * D_;

  // ---- Phase 0: slots, LN, bm, q ----
  for (int j = tid; j < 16 * 64; j += 512) {
    int row = j >> 6, c = j & 63;
    sl[j] = (row < S_) ? slots_g[((size_t)b * S_ + row) * 64 + c] : 0.f;
  }
  for (int j = tid; j < N_; j += 512) la[j] = la_g[(size_t)b * N_ + j];
  __syncthreads();
  if (tid < 16) {
    if (tid < S_) {
      const float* row = sl + tid * 64;
      float m = 0.f;
      for (int c = 0; c < 64; ++c) m += row[c];
      m *= (1.f / 64.f);
      float v = 0.f;
      for (int c = 0; c < 64; ++c) { float d = row[c] - m; v += d * d; }
      v *= (1.f / 64.f);
      float inv = rsqrtf(v + 1e-5f);
      for (int c = 0; c < 64; ++c)
        snb[tid * 66 + c] = (row[c] - m) * inv * lnsg[c] + lnsb[c];
    } else {
      for (int c = 0; c < 64; ++c) snb[tid * 66 + c] = 0.f;
    }
  }
  __syncthreads();
  if (tid < 16) {
    float lg = -3.4e38f;
    if (tid < S_) {
      lg = ws_b[0];
      for (int d = 0; d < 64; ++d) lg += snb[tid * 66 + d] * ws_w[d];
    }
    wt[tid] = lg;
  }
  __syncthreads();
  if (tid == 0) {
    float M = -3.4e38f;
    for (int s = 0; s < S_; ++s) M = fmaxf(M, wt[s]);
    float sm = 0.f;
    for (int s = 0; s < S_; ++s) sm += __expf(wt[s] - M);
    float lse = M + __logf(sm);
    float lS = __logf((float)S_);
    for (int s = 0; s < 16; ++s) {
      logb[s] = (s < S_) ? (wt[s] - lse + lS) : -1e30f;
      vh[s] = 0.f;  // dual warm-start v=0
    }
  }
  for (int o = tid; o < 16 * 64; o += 512) {  // q = sn @ Wq
    int row = o >> 6, c = o & 63;
    float acc = 0.f;
    for (int d = 0; d < 64; ++d) acc += snb[row * 66 + d] * Wq[d * 64 + c];
    qb[row * 66 + c] = acc;
  }
  __syncthreads();
  if (tid < 16) {
    float s2 = 0.f;
    for (int c = 0; c < 64; ++c) { float xx = qb[tid * 66 + c]; s2 += xx * xx; }
    q2[tid] = s2;
  }
  __syncthreads();

  // ---- Phase 1: p = -cdist(k,q), WMMA gram ----
  {
    v16bf qf0 = frag_b(qb, 66, 0);
    v16bf qf1 = frag_b(qb, 66, 32);
    const int ncol = lane & 15, hh = lane >> 4;
    const float q2s = q2[ncol];
    for (int nt = w; nt < 64; nt += 16) {
      const bf16* ab = kbg + (size_t)nt * 16 * 64;
      v16bf a0 = frag_a(ab, 64, 0);
      v16bf a1 = frag_a(ab, 64, 32);
      v8f acc = {};
      acc = wmma_bf16(a0, qf0, acc);
      acc = wmma_bf16(a1, qf1, acc);
#pragma unroll
      for (int r = 0; r < 8; ++r) {
        int n = nt * 16 + r + 8 * hh;
        float d2 = k2g[(size_t)b * N_ + n] + q2s - 2.f * acc[r];
        float Cv = sqrtf(fmaxf(d2, 1e-12f));
        p[n * 16 + ncol] = (ncol < S_) ? -Cv : -1e30f;  // finite pad: no NaN paths
      }
    }
  }
  __syncthreads();

  // ---- Phase 2: MESH (4x grad-of-entropy) + final Sinkhorn ----
  for (int mesh = 0; mesh < MESH_ITERS_; ++mesh) {
    sk_forward(p, uh, vh, la, logb);
    mesh_backward(p, gp, uh, vh, gu, gv, Ls, red);
    for (int j = tid; j < 16384; j += 512) p[j] -= gp[j];  // C -= lr*dH/dC (lr=1)
    if (tid < 16) vh[tid] = vh[5 * 16 + tid];              // warm-start duals
    __syncthreads();
  }
  sk_forward(p, uh, vh, la, logb);
  {  // T = exp(p + u5 + v5) into gp
    const float* u5 = uh + 4 * 1024;
    const float* v5 = vh + 5 * 16;
    for (int n = tid; n < N_; n += 512) {
      const float* pr = p + n * 16;
      float* gr = gp + n * 16;
      const float u = u5[n];
#pragma unroll
      for (int s = 0; s < 16; ++s) gr[s] = __expf(pr[s] + u + v5[s]);
    }
  }
  __syncthreads();

  // ---- Phase 3: attn out + updates = T^T @ v (WMMA, lockstep staging) ----
  for (int j = tid; j < S_ * N_; j += 512) {
    int s = j >> 10, n = j & 1023;
    attn_out[((size_t)b * S_ + s) * N_ + n] = gp[n * 16 + s];
  }
  v8f uacc = {};
  const int ct = w & 3;
  for (int kc = 0; kc < 32; ++kc) {
    {  // stage T^T (16 x 32) : exactly 512 elements / 512 threads
      int i = tid >> 4, s = tid & 15;
      stgT[s * 34 + i] = gp[(kc * 32 + i) * 16 + s];
    }
    for (int j = tid; j < 2048; j += 512) {  // stage v^T (64 x 32)
      int i = j >> 6, d = j & 63;
      stgV[d * 34 + i] = (float)vbg[(size_t)(kc * 32 + i) * 64 + d];
    }
    if (kc + 1 < 32)  // global_prefetch_b8 the next v chunk
      __builtin_prefetch((const void*)(vbg + (size_t)(kc + 1) * 32 * 64 + tid * 4), 0, 1);
    __syncthreads();
    if (w < 4) {
      v16bf at = frag_a(stgT, 34, 0);
      v16bf bt = frag_b(stgV + ct * 16 * 34, 34, 0);
      uacc = wmma_bf16(at, bt, uacc);
    }
    __syncthreads();
  }
  if (w < 4) {
#pragma unroll
    for (int r = 0; r < 8; ++r)
      upd[(r + 8 * (lane >> 4)) * 64 + ct * 16 + (lane & 15)] = uacc[r];
  }
  __syncthreads();

  // ---- Phase 4: GRU + MLP (11 slot rows) ----
  float* gi = uh;         // 11*192, reuse u-history
  float* gh = uh + 2112;  // 11*192
  for (int j = tid; j < S_ * 192; j += 512) {
    int row = j / 192, g = j % 192;
    float a1 = bih[g], a2 = bhh[g];
    const float* ur = upd + row * 64;
    const float* hr = sl + row * 64;
    for (int d = 0; d < 64; ++d) {
      a1 += ur[d] * Wih[d * 192 + g];
      a2 += hr[d] * Whh[d * 192 + g];
    }
    gi[j] = a1;
    gh[j] = a2;
  }
  __syncthreads();
  for (int j = tid; j < S_ * 64; j += 512) {
    int row = j >> 6, c = j & 63;
    float ir = gi[row * 192 + c], iz = gi[row * 192 + 64 + c], in_ = gi[row * 192 + 128 + c];
    float hrv = gh[row * 192 + c], hz = gh[row * 192 + 64 + c], hn = gh[row * 192 + 128 + c];
    float r = 1.f / (1.f + __expf(-(ir + hrv)));
    float z = 1.f / (1.f + __expf(-(iz + hz)));
    float nn = tanhf(in_ + r * hn);
    sl[row * 64 + c] = (1.f - z) * nn + z * sl[row * 64 + c];
  }
  __syncthreads();
  if (tid < S_) {  // ff layernorm into qb
    const float* row = sl + tid * 64;
    float m = 0.f;
    for (int c = 0; c < 64; ++c) m += row[c];
    m *= (1.f / 64.f);
    float v = 0.f;
    for (int c = 0; c < 64; ++c) { float d = row[c] - m; v += d * d; }
    v *= (1.f / 64.f);
    float inv = rsqrtf(v + 1e-5f);
    for (int c = 0; c < 64; ++c)
      qb[tid * 66 + c] = (row[c] - m) * inv * lnfg[c] + lnfb[c];
  }
  __syncthreads();
  float* relu1 = gp;  // reuse T buffer
  for (int j = tid; j < S_ * 128; j += 512) {
    int row = j >> 7, hc = j & 127;
    float a = fc1_b[hc];
    for (int d = 0; d < 64; ++d) a += qb[row * 66 + d] * fc1_w[d * 128 + hc];
    relu1[j] = fmaxf(a, 0.f);
  }
  __syncthreads();
  for (int j = tid; j < S_ * 64; j += 512) {
    int row = j >> 6, c = j & 63;
    float a = fc2_b[c];
    for (int hc = 0; hc < 128; ++hc) a += relu1[row * 128 + hc] * fc2_w[hc * 64 + c];
    slots_g[((size_t)b * S_ + row) * 64 + c] = sl[row * 64 + c] + a;
  }
}

// ---------------------------------------------------------------------------
extern "C" void kernel_launch(void* const* d_in, const int* in_sizes, int n_in,
                              void* d_out, int out_size, void* d_ws, size_t ws_size,
                              hipStream_t stream) {
  (void)in_sizes; (void)n_in; (void)out_size; (void)ws_size;
  const float* inputs     = (const float*)d_in[0];
  const float* slots_init = (const float*)d_in[1];
  const float* slots_mu   = (const float*)d_in[2];
  const float* slots_lsig = (const float*)d_in[3];
  const float* Wq   = (const float*)d_in[4];
  const float* Wk   = (const float*)d_in[5];
  const float* Wv   = (const float*)d_in[6];
  const float* Wih  = (const float*)d_in[7];
  const float* Whh  = (const float*)d_in[8];
  const float* bih  = (const float*)d_in[9];
  const float* bhh  = (const float*)d_in[10];
  const float* fc1_w = (const float*)d_in[11];
  const float* fc1_b = (const float*)d_in[12];
  const float* fc2_w = (const float*)d_in[13];
  const float* fc2_b = (const float*)d_in[14];
  const float* ln_in_g = (const float*)d_in[15];
  const float* ln_in_b = (const float*)d_in[16];
  const float* ln_sl_g = (const float*)d_in[17];
  const float* ln_sl_b = (const float*)d_in[18];
  const float* ln_ff_g = (const float*)d_in[19];
  const float* ln_ff_b = (const float*)d_in[20];
  const float* wi_w = (const float*)d_in[21];
  const float* wi_b = (const float*)d_in[22];
  const float* ws_w = (const float*)d_in[23];
  const float* ws_b = (const float*)d_in[24];

  char* ws = (char*)d_ws;
  bf16* k_bf    = (bf16*)(ws);                    // 32 MB
  bf16* v_bf    = (bf16*)(ws + 33554432ull);      // 32 MB
  float* k2     = (float*)(ws + 67108864ull);     // 1 MB
  float* logits = (float*)(ws + 68157440ull);     // 1 MB
  float* log_a  = (float*)(ws + 69206016ull);     // 1 MB
  float* slots  = (float*)(ws + 70254592ull);     // 704 KB

  float* out = (float*)d_out;
  float* attn_out = out + (size_t)B_ * S_ * D_;

  const size_t proj_smem = (128 * 66) * 4 + 2 * (64 * 66) * 2;  // 50,688 B
  const size_t iter_smem = (size_t)SH_TOT * 4;                  // 188,992 B

  proj_kernel<<<(B_ * N_) / 128, 256, proj_smem, stream>>>(
      inputs, Wk, Wv, ln_in_g, ln_in_b, wi_w, wi_b, k_bf, v_bf, logits);
  k2_kernel<<<(B_ * N_) / 256, 256, 0, stream>>>(k_bf, k2);
  softmax_a_kernel<<<B_, 256, 0, stream>>>(logits, log_a);
  slots_init_kernel<<<(B_ * S_ * D_ + 255) / 256, 256, 0, stream>>>(
      slots_init, slots_mu, slots_lsig, slots);
  for (int it = 0; it < ITERS_; ++it) {
    iter_kernel<<<B_, 512, iter_smem, stream>>>(
        k_bf, v_bf, k2, log_a, slots, attn_out, Wq, ln_sl_g, ln_sl_b, ws_w, ws_b,
        Wih, Whh, bih, bhh, fc1_w, fc1_b, fc2_w, fc2_b, ln_ff_g, ln_ff_b);
  }
  copy_kernel<<<(B_ * S_ * D_ + 255) / 256, 256, 0, stream>>>(slots, out);
}
